// SelfAttention_80341658239247
// MI455X (gfx1250) — matrix operand order repaired and checked by reading, not compiled, not run on hardware
//
#include <hip/hip_runtime.h>
#include <hip/hip_bf16.h>

// ---------------- problem constants ----------------
#define DIMD   768
#define NBATCH 256
#define NTOK   205
#define MTOT   (NBATCH * NTOK)      // 52480 tokens, divisible by 16
#define NC     8
#define HGRID  14
#define HW     196
#define NROWPAD 208                 // token rows padded to multiple of 16
#define NPADB   224                 // K padding for attn@v (7 x 32)
#define VT_STRIDE (NBATCH * NPADB)  // 57344, token stride of transposed v
#define ATTN_SCALE 0.03608439182435161f  // 768^-0.5

// TDM availability (tensor_load_to_lds arity differs across toolchains)
#if defined(__has_builtin)
# if __has_builtin(__builtin_amdgcn_tensor_load_to_lds) && \
     __has_builtin(__builtin_amdgcn_s_wait_tensorcnt)
#  define HAVE_TDM 1
# endif
#endif
#ifndef HAVE_TDM
# define HAVE_TDM 0
#endif

typedef __attribute__((ext_vector_type(16))) __bf16        v16bf;
typedef __attribute__((ext_vector_type(8)))  float         v8f;
typedef __attribute__((ext_vector_type(4)))  unsigned int  u32x4;
typedef __attribute__((ext_vector_type(4)))  unsigned int  v4u;
typedef __attribute__((ext_vector_type(8)))  int           v8i;
typedef __attribute__((ext_vector_type(4)))  int           v4i;

union FragB16 {            // one 16x32 (A) or 32x16 (B) bf16 operand per wave
    u32x4        q[2];
    unsigned int u[8];
    v16bf        v;
};

__device__ __forceinline__ unsigned short f2bf(float f) {
    unsigned int u = __float_as_uint(f);
    u += 0x7FFFu + ((u >> 16) & 1u);        // round-to-nearest-even
    return (unsigned short)(u >> 16);
}

__device__ __forceinline__ v8f wmma_bf16(const FragB16& a, const FragB16& b, v8f c) {
    return __builtin_amdgcn_wmma_f32_16x16x32_bf16(
        false, a.v, false, b.v, (short)0, c, false, false);
}

// =====================================================================
// Kernel 0: bulk f32 -> bf16 conversion (bandwidth bound, one pass).
// =====================================================================
__global__ __launch_bounds__(256)
void cvt_bf16_kernel(const float* __restrict__ src,
                     unsigned short* __restrict__ dst, int n8) {
    const int idx = blockIdx.x * blockDim.x + threadIdx.x;
    if (idx >= n8) return;
    const float* p = src + (size_t)idx * 8;
    u32x4 o;
    o.x = (unsigned)f2bf(p[0]) | ((unsigned)f2bf(p[1]) << 16);
    o.y = (unsigned)f2bf(p[2]) | ((unsigned)f2bf(p[3]) << 16);
    o.z = (unsigned)f2bf(p[4]) | ((unsigned)f2bf(p[5]) << 16);
    o.w = (unsigned)f2bf(p[6]) | ((unsigned)f2bf(p[7]) << 16);
    *(u32x4*)(dst + (size_t)idx * 8) = o;
}

// =====================================================================
// Kernel 1: q/k/v projections.  y[m,e] = sum_d x[m,d] * W[e,d]
//   Pure b128 loads + wmma (operands pre-converted to bf16).
//   z=0 -> q, z=1 -> k, z=2 -> v stored TRANSPOSED: vT[e][b*224 + n].
// =====================================================================
__global__ __launch_bounds__(128)
void qkv_proj_kernel(const unsigned short* __restrict__ xbf,
                     const unsigned short* __restrict__ wbf,   // [3][768][768]
                     unsigned short* __restrict__ qws,
                     unsigned short* __restrict__ kws,
                     unsigned short* __restrict__ vtws) {
    const int lane = threadIdx.x & 31;
    const int wave = threadIdx.x >> 5;
    const int lrow = lane & 15;
    const int hi   = lane >> 4;
    const int m0   = blockIdx.x * 16;                    // token tile base
    const int e0   = blockIdx.y * 256 + wave * 64;       // output-feature base
    const int z    = blockIdx.z;

    const unsigned short* xrow = xbf + (size_t)(m0 + lrow) * DIMD;
    const unsigned short* wz   = wbf + (size_t)z * DIMD * DIMD;

    v8f acc[4] = {};
    for (int kk = 0; kk < DIMD; kk += 32) {
        __builtin_prefetch(xrow + kk + 128, 0, 1);       // global_prefetch_b8
        FragB16 a;
        a.q[0] = *(const u32x4*)(xrow + kk + hi * 16);
        a.q[1] = *(const u32x4*)(xrow + kk + hi * 16 + 8);
#pragma unroll
        for (int t = 0; t < 4; ++t) {
            const unsigned short* wr =
                wz + (size_t)(e0 + t * 16 + lrow) * DIMD + kk + hi * 16;
            FragB16 bfr;
            bfr.q[0] = *(const u32x4*)(wr);
            bfr.q[1] = *(const u32x4*)(wr + 8);
            acc[t] = wmma_bf16(a, bfr, acc[t]);
        }
    }

    if (z < 2) {
        unsigned short* out = (z == 0) ? qws : kws;
#pragma unroll
        for (int t = 0; t < 4; ++t) {
            const int e = e0 + t * 16 + lrow;
#pragma unroll
            for (int r = 0; r < 8; ++r) {
                const int m = m0 + r + hi * 8;
                out[(size_t)m * DIMD + e] = f2bf(acc[t][r]);
            }
        }
    } else {
#pragma unroll
        for (int t = 0; t < 4; ++t) {
            const int e = e0 + t * 16 + lrow;
#pragma unroll
            for (int r = 0; r < 8; ++r) {
                const int m = m0 + r + hi * 8;
                const int b = m / NTOK;
                const int n = m - b * NTOK;
                vtws[(size_t)e * VT_STRIDE + (size_t)b * NPADB + n] = f2bf(acc[t][r]);
            }
        }
    }
}

// =====================================================================
// Kernel 2: per-batch scores = q @ k^T into LDS, gaussian mask on the
// concept-token rows, row softmax(SCALE * s).
//   grid = (256 batches, 2 row-halves: 112 / 96 rows), block = 256 (8 waves)
// =====================================================================
__global__ __launch_bounds__(256)
void attn_softmax_kernel(const unsigned short* __restrict__ qws,
                         const unsigned short* __restrict__ kws,
                         float* __restrict__ attn_out,
                         unsigned short* __restrict__ attn_bf) {
    __shared__ float smem[112 * NROWPAD];   // up to 112 rows x 208 score cols
    __shared__ float red[256];              // per-wave reduction scratch

    const int b       = blockIdx.x;
    const int hhalf   = blockIdx.y;
    const int rowbase = hhalf * 112;
    const int nitile  = hhalf ? 6 : 7;
    const int nrows   = nitile * 16;
    const int ntiles  = nitile * 13;

    const int lane = threadIdx.x & 31;
    const int wave = threadIdx.x >> 5;
    const int lrow = lane & 15;
    const int hi   = lane >> 4;
    const size_t tokbase = (size_t)b * NTOK;

    // ---------------- phase 1: score GEMM tiles ----------------
    for (int t = wave; t < ntiles; t += 8) {
        const int i = t / 13;
        const int j = t % 13;
        const int grow = rowbase + i * 16 + lrow;   // query row (0..207)
        const int kcol = j * 16 + lrow;             // key row   (0..207)
        const bool qv = grow < NTOK;
        const bool kv = kcol < NTOK;
        const unsigned short* qp = qws + (tokbase + grow) * DIMD;
        const unsigned short* kp = kws + (tokbase + kcol) * DIMD;

        v8f acc = {};
        for (int kk = 0; kk < DIMD; kk += 32) {
            FragB16 a, bb;
            if (qv) {
                a.q[0] = *(const u32x4*)(qp + kk + hi * 16);
                a.q[1] = *(const u32x4*)(qp + kk + hi * 16 + 8);
            } else {
#pragma unroll
                for (int h = 0; h < 8; ++h) a.u[h] = 0u;
            }
            if (kv) {
                bb.q[0] = *(const u32x4*)(kp + kk + hi * 16);
                bb.q[1] = *(const u32x4*)(kp + kk + hi * 16 + 8);
            } else {
#pragma unroll
                for (int h = 0; h < 8; ++h) bb.u[h] = 0u;
            }
            acc = wmma_bf16(a, bb, acc);
        }
#pragma unroll
        for (int r = 0; r < 8; ++r)
            smem[(i * 16 + r + hi * 8) * NROWPAD + j * 16 + lrow] = acc[r];
    }
    __syncthreads();

    // ------- phase 2: gaussian mask (rows 197..204, cols 1..196) -------
    if (hhalf == 1 && threadIdx.x < NC) {
        const int lr = (197 - 112) + threadIdx.x;   // local row 85..92
        float* row = smem + lr * NROWPAD;
        float mx = row[1];
        int   loc = 0;
        for (int f = 1; f < HW; ++f) {              // first-max (argmax) semantics
            const float vv = row[1 + f];
            if (vv > mx) { mx = vv; loc = f; }
        }
        const float cx = (float)(loc % HGRID);      // max_x (compared against rows)
        const float cy = (float)(loc / HGRID);      // max_y (compared against cols)
        for (int f = 0; f < HW; ++f) {
            const float dr = (float)(f / HGRID) - cx;
            const float dc = (float)(f % HGRID) - cy;
            row[1 + f] *= __expf(-2.0f * (dr * dr + dc * dc));  // 1/(2*0.5^2) = 2
        }
    }
    __syncthreads();

    // ---------------- phase 3: softmax(SCALE * s) per row ----------------
    for (int rl = wave; rl < nrows; rl += 8) {
        const int gr = rowbase + rl;
        unsigned short* abf = attn_bf + ((size_t)b * NROWPAD + gr) * NPADB;
        if (gr < NTOK) {
            const float* row = smem + rl * NROWPAD;
            float m = -1e30f;
            for (int c = lane; c < NTOK; c += 32)
                m = fmaxf(m, row[c] * ATTN_SCALE);
            red[wave * 32 + lane] = m;
            float mm = -1e30f;
#pragma unroll
            for (int l = 0; l < 32; ++l) mm = fmaxf(mm, red[wave * 32 + l]);

            float s = 0.0f;
            for (int c = lane; c < NTOK; c += 32)
                s += __expf(row[c] * ATTN_SCALE - mm);
            red[wave * 32 + lane] = s;
            float ss = 0.0f;
#pragma unroll
            for (int l = 0; l < 32; ++l) ss += red[wave * 32 + l];
            const float inv = 1.0f / ss;

            float* aout = attn_out + ((size_t)b * NTOK + gr) * NTOK;
            for (int c = lane; c < NPADB; c += 32) {
                float p = 0.0f;
                if (c < NTOK) {
                    p = __expf(row[c] * ATTN_SCALE - mm) * inv;
                    aout[c] = p;
                }
                abf[c] = f2bf(p);          // padded cols -> exact 0
            }
        } else {
            for (int c = lane; c < NPADB; c += 32) abf[c] = 0;  // padded rows
        }
    }
}

// =====================================================================
// Kernel 3: out[b,qi,e] = sum_kj attn[b,qi,kj] * v[b,kj,e]
//   One block per (256-col chunk, batch). The 256x224 bf16 vT panel is
//   staged ONCE into LDS via the Tensor Data Mover (tensor_load_to_lds,
//   s_wait_tensorcnt), then all 13 row tiles are computed from LDS.
//   grid = (3 col chunks, 256 batches), block = 128 (4 waves).
// =====================================================================
__global__ __launch_bounds__(128)
void out_gemm_kernel(const unsigned short* __restrict__ attn_bf,
                     const unsigned short* __restrict__ vtws,
                     float* __restrict__ outp) {
#if HAVE_TDM
    __shared__ unsigned short lds_vt[256 * NPADB];   // 114,688 B
#endif
    const int lane  = threadIdx.x & 31;
    const int wave  = threadIdx.x >> 5;
    const int lrow  = lane & 15;
    const int hi    = lane >> 4;
    const int e0blk = blockIdx.x * 256;
    const int b     = blockIdx.y;
    const int ewave = wave * 64;

#if HAVE_TDM
    if (wave == 0) {
        // ---- Tensor DMA descriptor (D#), 2D tile: 224 x 256 rows, bf16 ----
        const unsigned long long gaddr = (unsigned long long)(size_t)
            (vtws + (size_t)e0blk * VT_STRIDE + (size_t)b * NPADB);
        const unsigned lds_off = (unsigned)(size_t)(&lds_vt[0]);
        v4u g0;
        g0.x = 0x1u;                                  // count=1, user mode
        g0.y = lds_off;                               // LDS byte address
        g0.z = (unsigned)(gaddr & 0xFFFFFFFFu);       // global_addr[31:0]
        g0.w = (unsigned)((gaddr >> 32) & 0x1FFFFFFu) // global_addr[56:32]
               | (2u << 30);                          // type = 2 ("image")
        v8i g1;
        g1[0] = (int)(1u << 16);          // wg_mask=0, data_size=1 (2 bytes)
        g1[1] = (int)(224u << 16);        // tensor_dim0 = 224 (low 16)
        g1[2] = (int)(256u << 16);        // tensor_dim0 hi=0 | tensor_dim1=256
        g1[3] = (int)(224u << 16);        // tensor_dim1 hi=0 | tile_dim0=224
        g1[4] = 256;                      // tile_dim1=256, tile_dim2=0
        g1[5] = (int)VT_STRIDE;           // tensor_dim0_stride = 57344
        g1[6] = 0;                        // stride hi | tensor_dim1_stride lo
        g1[7] = 0;
        v4i gz = {0, 0, 0, 0};
#if __clang_major__ >= 23
        v8i gz8 = {0, 0, 0, 0, 0, 0, 0, 0};
        __builtin_amdgcn_tensor_load_to_lds(g0, g1, gz, gz, gz8, 0);
#else
        __builtin_amdgcn_tensor_load_to_lds(g0, g1, gz, gz, 0);
#endif
        __builtin_amdgcn_s_wait_tensorcnt(0);
    }
    __syncthreads();
#endif

    for (int i = 0; i < 13; ++i) {
        const unsigned short* ap =
            attn_bf + ((size_t)b * NROWPAD + i * 16 + lrow) * NPADB;
        v8f acc[4] = {};
        for (int kk = 0; kk < NPADB; kk += 32) {
            FragB16 a;
            a.q[0] = *(const u32x4*)(ap + kk + hi * 16);
            a.q[1] = *(const u32x4*)(ap + kk + hi * 16 + 8);
#pragma unroll
            for (int t = 0; t < 4; ++t) {
                FragB16 bb;
#if HAVE_TDM
                const unsigned short* vp =
                    lds_vt + (size_t)(ewave + t * 16 + lrow) * NPADB + kk + hi * 16;
#else
                const unsigned short* vp =
                    vtws + (size_t)(e0blk + ewave + t * 16 + lrow) * VT_STRIDE +
                    (size_t)b * NPADB + kk + hi * 16;
#endif
                bb.q[0] = *(const u32x4*)(vp);
                bb.q[1] = *(const u32x4*)(vp + 8);
                acc[t] = wmma_bf16(a, bb, acc[t]);
            }
        }
#pragma unroll
        for (int t = 0; t < 4; ++t) {
            const int e = e0blk + ewave + t * 16 + lrow;
#pragma unroll
            for (int r = 0; r < 8; ++r) {
                const int qi = i * 16 + r + hi * 8;
                if (qi < NTOK)
                    outp[((size_t)b * NTOK + qi) * DIMD + e] = acc[t][r];
            }
        }
    }
}

// =====================================================================
extern "C" void kernel_launch(void* const* d_in, const int* in_sizes, int n_in,
                              void* d_out, int out_size, void* d_ws, size_t ws_size,
                              hipStream_t stream) {
    const float* x  = (const float*)d_in[0];
    const float* Wq = (const float*)d_in[1];
    const float* Wk = (const float*)d_in[2];
    const float* Wv = (const float*)d_in[3];

    // d_out: attn [B*N*N] f32, then out [B*N*D] f32
    float* attn_out = (float*)d_out;
    float* outp     = attn_out + (size_t)NBATCH * NTOK * NTOK;

    // workspace layout (bytes):
    //   [0]           x_bf16 [52480][768]     80,609,280  (dead after k1)
    //                 attn_bf [256][208][224] 23,855,104  (aliases x_bf16)
    //   [80,609,280]  w_bf16 [3][768][768]     3,538,944
    //   [84,148,224]  q bf16 [52480][768]     80,609,280
    //   [164,757,504] k bf16 [52480][768]     80,609,280
    //   [245,366,784] vT bf16 [768][57344]    88,080,384
    //   total: 333,447,168
    char* ws = (char*)d_ws;
    const size_t x_bytes  = (size_t)MTOT * DIMD * sizeof(unsigned short);
    const size_t w_bytes  = (size_t)3 * DIMD * DIMD * sizeof(unsigned short);
    unsigned short* xbf  = (unsigned short*)(ws);
    unsigned short* abf  = (unsigned short*)(ws);              // alias (post-k1)
    unsigned short* wbf  = (unsigned short*)(ws + x_bytes);
    unsigned short* qws  = (unsigned short*)(ws + x_bytes + w_bytes);
    unsigned short* kws  = (unsigned short*)(ws + x_bytes + w_bytes + x_bytes);
    unsigned short* vtws = (unsigned short*)(ws + x_bytes + w_bytes + 2 * x_bytes);

    // 0) one-shot f32 -> bf16 conversions (x and the three weight matrices)
    {
        const int n8x = MTOT * DIMD / 8;          // 5,038,080
        const int n8w = DIMD * DIMD / 8;          // 73,728
        cvt_bf16_kernel<<<(n8x + 255) / 256, 256, 0, stream>>>(x, xbf, n8x);
        cvt_bf16_kernel<<<(n8w + 255) / 256, 256, 0, stream>>>(Wq, wbf, n8w);
        cvt_bf16_kernel<<<(n8w + 255) / 256, 256, 0, stream>>>(Wk, wbf + (size_t)DIMD * DIMD, n8w);
        cvt_bf16_kernel<<<(n8w + 255) / 256, 256, 0, stream>>>(Wv, wbf + (size_t)2 * DIMD * DIMD, n8w);
    }

    // 1) projections: grid (52480/16 M tiles, 768/256 col chunks, 3 matrices)
    qkv_proj_kernel<<<dim3(MTOT / 16, DIMD / 256, 3), 128, 0, stream>>>(
        xbf, wbf, qws, kws, vtws);

    // 2) fused scores + mask + softmax: (256 batches, 2 row halves)
    attn_softmax_kernel<<<dim3(NBATCH, 2), 256, 0, stream>>>(
        qws, kws, attn_out, abf);

    // 3) out = attn @ v: (3 col chunks, 256 batches), vT panel via TDM -> LDS
    out_gemm_kernel<<<dim3(3, NBATCH), 128, 0, stream>>>(
        abf, vtws, outp);
}
